// MoE_28879360098375
// MI455X (gfx1250) — compile-verified, matrix operand order checked by reading, not run on hardware
//
#include <hip/hip_runtime.h>

// ---------------------------------------------------------------------------
// MoE (top-2 of 6 routed experts + shared expert), DIM=INTER=1024, T=32768.
// bf16 WMMA grouped GEMMs over gathered per-expert token lists; shared expert
// is "expert 6" with identity routing (gate=1). Double-buffered LDS staging
// via GLOBAL_LOAD_ASYNC_TO_LDS_B128 when the toolchain exposes it.
// ---------------------------------------------------------------------------

#define T_TOKENS  (8 * 4096)
#define DIM       1024
#define INTER     1024
#define NEXP      6
#define NEXP_ALL  7                    // 6 routed + shared
#define CAP       (3 * T_TOKENS)       // 2T routed slots + T shared slots

#define LDA 40                         // padded LDS stride (halfs): BK=32 + 8
#define LDB 40

typedef __attribute__((ext_vector_type(16))) __bf16 v16bf;
typedef __attribute__((ext_vector_type(8)))  float  v8f;
typedef int v4i_raw __attribute__((vector_size(16)));           // b128 payload
typedef __attribute__((address_space(1))) v4i_raw v4i_glob;     // global AS
typedef __attribute__((address_space(3))) v4i_raw v4i_lds;      // LDS AS

#if defined(__AMDGCN__) && __has_builtin(__builtin_amdgcn_global_load_async_to_lds_b128)
#define HAS_ASYNC 1
#else
#define HAS_ASYNC 0
#endif

#if HAS_ASYNC
__device__ __forceinline__ void stage16(unsigned short* dst, const unsigned short* src) {
    __builtin_amdgcn_global_load_async_to_lds_b128(
        (v4i_glob*)const_cast<unsigned short*>(src),
        (v4i_lds*)dst, 0, 0);
}
__device__ __forceinline__ void wait_async() {
#if __has_builtin(__builtin_amdgcn_s_wait_asynccnt)
    __builtin_amdgcn_s_wait_asynccnt(0);
#else
    asm volatile("s_wait_asynccnt 0" ::: "memory");
#endif
}
#else
__device__ __forceinline__ void stage16(unsigned short* dst, const unsigned short* src) {
    *(uint4*)dst = *(const uint4*)src;
}
__device__ __forceinline__ void wait_async() {}
#endif

__device__ __forceinline__ unsigned short f2bf(float f) {
    unsigned int u = __float_as_uint(f);
    u += 0x7FFFu + ((u >> 16) & 1u);   // round-to-nearest-even
    return (unsigned short)(u >> 16);
}

__device__ __forceinline__ float fast_silu(float g) {
    // g * sigmoid(g) via v_exp_f32 + v_rcp_f32 (avoid IEEE divide sequence)
    return g * __builtin_amdgcn_rcpf(1.f + __expf(-g));
}

// ----------------------------- small utility kernels -----------------------

__global__ void k_cvt_bf16(const float* __restrict__ s,
                           unsigned short* __restrict__ d, int n) {
    int i = (blockIdx.x * 256 + threadIdx.x) * 4;
    if (i >= n) return;
    float4 v = *(const float4*)(s + i);
    *(ushort4*)(d + i) = make_ushort4(f2bf(v.x), f2bf(v.y), f2bf(v.z), f2bf(v.w));
}

__global__ void k_zero_f4(float* __restrict__ p, int n) {
    int i = (blockIdx.x * 256 + threadIdx.x) * 4;
    if (i < n) *(float4*)(p + i) = make_float4(0.f, 0.f, 0.f, 0.f);
}

__global__ void k_zero_i(int* __restrict__ p, int n) {
    int i = blockIdx.x * 32 + threadIdx.x;
    if (i < n) p[i] = 0;
}

// ----------------------------- gate: sigmoid + top-2 -----------------------
__global__ __launch_bounds__(256) void k_gate(
    const float* __restrict__ x, const float* __restrict__ gw,
    const float* __restrict__ gb,
    int* __restrict__ tki, float* __restrict__ tkw, int* __restrict__ counts) {
    int t = blockIdx.x * 8 + (threadIdx.x >> 5);
    int lane = threadIdx.x & 31;
    if (t >= T_TOKENS) return;
    const float* xr = x + (size_t)t * DIM;
    float acc[NEXP] = {0.f, 0.f, 0.f, 0.f, 0.f, 0.f};
    for (int k = lane; k < DIM; k += 32) {
        float xv = xr[k];
#pragma unroll
        for (int e = 0; e < NEXP; ++e) acc[e] += xv * gw[e * DIM + k];
    }
#pragma unroll
    for (int e = 0; e < NEXP; ++e)
#pragma unroll
        for (int off = 16; off; off >>= 1) acc[e] += __shfl_xor(acc[e], off);
    if (lane == 0) {
        float p[NEXP];
#pragma unroll
        for (int e = 0; e < NEXP; ++e)
            p[e] = __builtin_amdgcn_rcpf(1.f + __expf(-(acc[e] + gb[e])));
        int i0 = 0;
#pragma unroll
        for (int e = 1; e < NEXP; ++e) if (p[e] > p[i0]) i0 = e;
        int i1 = (i0 == 0) ? 1 : 0;
#pragma unroll
        for (int e = 0; e < NEXP; ++e) if (e != i0 && p[e] > p[i1]) i1 = e;
        float s = p[i0] + p[i1] + 1e-8f;
        float rs = __builtin_amdgcn_rcpf(s);
        tki[t * 2] = i0;      tkw[t * 2] = p[i0] * rs;
        tki[t * 2 + 1] = i1;  tkw[t * 2 + 1] = p[i1] * rs;
        atomicAdd(&counts[i0], 1);
        atomicAdd(&counts[i1], 1);
    }
}

__global__ void k_prefix(const int* __restrict__ counts,
                         int* __restrict__ offsets, int* __restrict__ widx) {
    if (threadIdx.x == 0 && blockIdx.x == 0) {
        int acc = 0;
        for (int e = 0; e < NEXP; ++e) {
            offsets[e] = acc; widx[e] = acc; acc += counts[e];
        }
        offsets[NEXP] = acc;                    // shared-expert base (== 2T)
        widx[NEXP] = acc;
        offsets[NEXP + 1] = acc + T_TOKENS;
    }
}

__global__ __launch_bounds__(256) void k_scatter(
    const int* __restrict__ tki, const float* __restrict__ tkw,
    const int* __restrict__ offsets, int* __restrict__ widx,
    int* __restrict__ rows, float* __restrict__ gv) {
    int t = blockIdx.x * 256 + threadIdx.x;
    if (t >= T_TOKENS) return;
#pragma unroll
    for (int k = 0; k < 2; ++k) {
        int e = tki[t * 2 + k];
        int pos = atomicAdd(&widx[e], 1);       // absolute slot index
        rows[pos] = t;
        gv[pos] = tkw[t * 2 + k];
    }
    int sb = offsets[NEXP];                     // shared expert: identity
    rows[sb + t] = t;
    gv[sb + t] = 1.0f;
}

// ----------------------------- GEMM1: H = silu(X W1^T) * (X W3^T) ----------
// 128x64 tile, 256 threads (8 waves as 4x2), 32x32 per wave (2x2 WMMA),
// BK=32, double-buffered LDS with async global->LDS prefetch.
__global__ __launch_bounds__(256) void k_gemm1(
    const unsigned short* __restrict__ Xb,     // [T, DIM] bf16
    const unsigned short* __restrict__ Wb1,    // [7, INTER, DIM] bf16
    const unsigned short* __restrict__ Wb3,    // [7, INTER, DIM] bf16
    const int* __restrict__ rows, const int* __restrict__ offsets,
    unsigned short* __restrict__ H)            // [CAP, INTER] bf16
{
    int e = blockIdx.z;
    int base = offsets[e];
    int cnt  = offsets[e + 1] - base;
    int m0 = blockIdx.y * 128;
    if (m0 >= cnt) return;
    int n0 = blockIdx.x * 64;
    const unsigned short* w1 = Wb1 + (size_t)e * INTER * DIM;
    const unsigned short* w3 = Wb3 + (size_t)e * INTER * DIM;

    __shared__ unsigned short sA[2][128 * LDA];
    __shared__ unsigned short sB1[2][64 * LDB];
    __shared__ unsigned short sB3[2][64 * LDB];
    __shared__ int sRow[128];

    int tid = threadIdx.x;
    if (tid < 128) {
        int mm = m0 + tid; if (mm >= cnt) mm = cnt - 1;
        sRow[tid] = rows[base + mm];
    }
    __syncthreads();

    int lane = tid & 31, wid = tid >> 5;
    int wm = (wid >> 1) * 32, wn = (wid & 1) * 32;

    v8f acc1[2][2] = {};
    v8f acc3[2][2] = {};

    auto stage = [&](int kk, int buf) {
        // A tile 128x32: 512 16B-chunks, 2 per thread
#pragma unroll
        for (int it = 0; it < 2; ++it) {
            int c = tid + it * 256;
            int r = c >> 2, col = (c & 3) * 8;
            stage16(&sA[buf][r * LDA + col],
                    Xb + (size_t)sRow[r] * DIM + kk + col);
        }
        // B tiles 64x32: 256 16B-chunks, 1 per thread each
        int r = tid >> 2, col = (tid & 3) * 8;
        stage16(&sB1[buf][r * LDB + col], w1 + (size_t)(n0 + r) * DIM + kk + col);
        stage16(&sB3[buf][r * LDB + col], w3 + (size_t)(n0 + r) * DIM + kk + col);
    };

    stage(0, 0);
    wait_async();
    __syncthreads();

    for (int kk = 0; kk < DIM; kk += 32) {
        int buf = (kk >> 5) & 1;
        if (kk + 32 < DIM) stage(kk + 32, buf ^ 1);   // prefetch next chunk

        int koff = (lane >> 4) * 16;                   // lanes 16-31: K half 2
        int rl = lane & 15;
        v16bf a[2], b1f[2], b3f[2];
#pragma unroll
        for (int tm = 0; tm < 2; ++tm)
            a[tm] = *(const v16bf*)&sA[buf][(wm + tm * 16 + rl) * LDA + koff];
#pragma unroll
        for (int tn = 0; tn < 2; ++tn) {
            b1f[tn] = *(const v16bf*)&sB1[buf][(wn + tn * 16 + rl) * LDB + koff];
            b3f[tn] = *(const v16bf*)&sB3[buf][(wn + tn * 16 + rl) * LDB + koff];
        }
#pragma unroll
        for (int tm = 0; tm < 2; ++tm)
#pragma unroll
            for (int tn = 0; tn < 2; ++tn) {
                acc1[tm][tn] = __builtin_amdgcn_wmma_f32_16x16x32_bf16(
                    false, a[tm], false, b1f[tn], (short)0, acc1[tm][tn], false, false);
                acc3[tm][tn] = __builtin_amdgcn_wmma_f32_16x16x32_bf16(
                    false, a[tm], false, b3f[tn], (short)0, acc3[tm][tn], false, false);
            }
        wait_async();
        __syncthreads();
    }

    // epilogue: h = silu(g) * u. D layout: col = lane[3:0], row = i + 8*lane[4]
    int nl = lane & 15, mh = (lane >> 4) * 8;
#pragma unroll
    for (int tm = 0; tm < 2; ++tm)
#pragma unroll
        for (int tn = 0; tn < 2; ++tn)
#pragma unroll
            for (int i = 0; i < 8; ++i) {
                int m = wm + tm * 16 + mh + i;
                if (m0 + m >= cnt) continue;
                int n = wn + tn * 16 + nl;
                float hv = fast_silu(acc1[tm][tn][i]) * acc3[tm][tn][i];
                H[(size_t)(base + m0 + m) * INTER + n0 + n] = f2bf(hv);
            }
}

// ----------------------------- GEMM2: out[t] += gate * (H W2^T) ------------
__global__ __launch_bounds__(256) void k_gemm2(
    const unsigned short* __restrict__ H,      // [CAP, INTER] bf16
    const unsigned short* __restrict__ Wb2,    // [7, DIM, INTER] bf16
    const int* __restrict__ rows, const float* __restrict__ gv,
    const int* __restrict__ offsets,
    float* __restrict__ out)                   // [T, DIM] f32
{
    int e = blockIdx.z;
    int base = offsets[e];
    int cnt  = offsets[e + 1] - base;
    int m0 = blockIdx.y * 128;
    if (m0 >= cnt) return;
    int n0 = blockIdx.x * 64;
    const unsigned short* w2 = Wb2 + (size_t)e * DIM * INTER;

    __shared__ unsigned short sA[2][128 * LDA];
    __shared__ unsigned short sB[2][64 * LDB];

    int tid = threadIdx.x;
    int lane = tid & 31, wid = tid >> 5;
    int wm = (wid >> 1) * 32, wn = (wid & 1) * 32;

    v8f acc[2][2] = {};

    auto stage = [&](int kk, int buf) {
#pragma unroll
        for (int it = 0; it < 2; ++it) {
            int c = tid + it * 256;
            int r = c >> 2, col = (c & 3) * 8;
            int rr = m0 + r; if (rr >= cnt) rr = cnt - 1;
            stage16(&sA[buf][r * LDA + col],
                    H + (size_t)(base + rr) * INTER + kk + col);
        }
        int r = tid >> 2, col = (tid & 3) * 8;
        stage16(&sB[buf][r * LDB + col], w2 + (size_t)(n0 + r) * INTER + kk + col);
    };

    stage(0, 0);
    wait_async();
    __syncthreads();

    for (int kk = 0; kk < INTER; kk += 32) {
        int buf = (kk >> 5) & 1;
        if (kk + 32 < INTER) stage(kk + 32, buf ^ 1);

        int koff = (lane >> 4) * 16;
        int rl = lane & 15;
        v16bf a[2], b[2];
#pragma unroll
        for (int tm = 0; tm < 2; ++tm)
            a[tm] = *(const v16bf*)&sA[buf][(wm + tm * 16 + rl) * LDA + koff];
#pragma unroll
        for (int tn = 0; tn < 2; ++tn)
            b[tn] = *(const v16bf*)&sB[buf][(wn + tn * 16 + rl) * LDB + koff];
#pragma unroll
        for (int tm = 0; tm < 2; ++tm)
#pragma unroll
            for (int tn = 0; tn < 2; ++tn)
                acc[tm][tn] = __builtin_amdgcn_wmma_f32_16x16x32_bf16(
                    false, a[tm], false, b[tn], (short)0, acc[tm][tn], false, false);
        wait_async();
        __syncthreads();
    }

    int nl = lane & 15, mh = (lane >> 4) * 8;
#pragma unroll
    for (int tm = 0; tm < 2; ++tm)
#pragma unroll
        for (int tn = 0; tn < 2; ++tn)
#pragma unroll
            for (int i = 0; i < 8; ++i) {
                int m = wm + tm * 16 + mh + i;
                if (m0 + m >= cnt) continue;
                int slot = base + m0 + m;
                int t = rows[slot];
                float g = gv[slot];
                int n = wn + tn * 16 + nl;
                atomicAdd(&out[(size_t)t * DIM + n0 + n], g * acc[tm][tn][i]);
            }
}

// ----------------------------- host-side launch -----------------------------

extern "C" void kernel_launch(void* const* d_in, const int* in_sizes, int n_in,
                              void* d_out, int out_size, void* d_ws, size_t ws_size,
                              hipStream_t stream) {
    const float* x   = (const float*)d_in[0];
    const float* gw  = (const float*)d_in[1];
    const float* gb  = (const float*)d_in[2];
    const float* ew1 = (const float*)d_in[3];
    const float* ew2 = (const float*)d_in[4];
    const float* ew3 = (const float*)d_in[5];
    const float* fc1 = (const float*)d_in[6];
    const float* fc2 = (const float*)d_in[7];
    const float* fc3 = (const float*)d_in[8];
    float* out = (float*)d_out;

    const size_t MAT = (size_t)INTER * DIM;     // 1M elems per weight matrix

    char* p = (char*)d_ws;
    unsigned short* Wb1 = (unsigned short*)p;  p += NEXP_ALL * MAT * 2;
    unsigned short* Wb3 = (unsigned short*)p;  p += NEXP_ALL * MAT * 2;
    unsigned short* Wb2 = (unsigned short*)p;  p += NEXP_ALL * MAT * 2;
    unsigned short* Xb  = (unsigned short*)p;  p += (size_t)T_TOKENS * DIM * 2;
    unsigned short* H   = (unsigned short*)p;  p += (size_t)CAP * INTER * 2;
    int*   tki     = (int*)p;                  p += (size_t)T_TOKENS * 2 * 4;
    float* tkw     = (float*)p;                p += (size_t)T_TOKENS * 2 * 4;
    int*   counts  = (int*)p;                  p += 16 * 4;
    int*   offsets = (int*)p;                  p += 16 * 4;
    int*   widx    = (int*)p;                  p += 16 * 4;
    int*   rows    = (int*)p;                  p += (size_t)CAP * 4;
    float* gv      = (float*)p;                p += (size_t)CAP * 4;
    (void)ws_size; (void)n_in; (void)in_sizes; (void)out_size;

    auto cvt = [&](const float* s, unsigned short* d, size_t n) {
        int blocks = (int)((n / 4 + 255) / 256);
        k_cvt_bf16<<<blocks, 256, 0, stream>>>(s, d, (int)n);
    };
    // bf16 weight pack: experts 0..5 routed, expert 6 = shared (fc1/fc2/fc3)
    cvt(ew1, Wb1, NEXP * MAT);  cvt(fc1, Wb1 + NEXP * MAT, MAT);
    cvt(ew3, Wb3, NEXP * MAT);  cvt(fc2, Wb3 + NEXP * MAT, MAT);
    cvt(ew2, Wb2, NEXP * MAT);  cvt(fc3, Wb2 + NEXP * MAT, MAT);
    cvt(x, Xb, (size_t)T_TOKENS * DIM);

    k_zero_i<<<1, 32, 0, stream>>>(counts, 16);
    k_gate<<<T_TOKENS / 8, 256, 0, stream>>>(x, gw, gb, tki, tkw, counts);
    k_prefix<<<1, 1, 0, stream>>>(counts, offsets, widx);
    k_scatter<<<T_TOKENS / 256, 256, 0, stream>>>(tki, tkw, offsets, widx, rows, gv);

    k_zero_f4<<<(T_TOKENS * DIM / 4 + 255) / 256, 256, 0, stream>>>(out, T_TOKENS * DIM);

    dim3 g1(INTER / 64, (T_TOKENS + 127) / 128, NEXP_ALL);
    k_gemm1<<<g1, 256, 0, stream>>>(Xb, Wb1, Wb3, rows, offsets, H);

    dim3 g2(DIM / 64, (T_TOKENS + 127) / 128, NEXP_ALL);
    k_gemm2<<<g2, 256, 0, stream>>>(H, Wb2, rows, gv, offsets, out);
}
